// GraphSAGE_5342939316743
// MI455X (gfx1250) — matrix operand order repaired and checked by reading, not compiled, not run on hardware
//
#include <hip/hip_runtime.h>
#include <hip/hip_bf16.h>

// ---------------- problem constants (match reference) ----------------
#define NN   100000
#define EE   1600000
#define DIN  128
#define DHID 256
#define DOUT 64

// ---------------- vector types (trivial ext vectors) -----------------
typedef __attribute__((ext_vector_type(16))) __bf16        v16bf;
typedef __attribute__((ext_vector_type(8)))  float         v8f;
typedef __attribute__((ext_vector_type(16))) unsigned short v16us;
typedef __attribute__((ext_vector_type(8)))  unsigned short v8us;
typedef __attribute__((ext_vector_type(4)))  unsigned short v4us;
typedef __attribute__((ext_vector_type(4)))  float         v4f;

// fp32 -> bf16 (round to nearest even), raw u16
static __device__ __forceinline__ unsigned short f2bf(float f) {
    unsigned int u = __float_as_uint(f);
    u += 0x7FFFu + ((u >> 16) & 1u);
    return (unsigned short)(u >> 16);
}
static __device__ __forceinline__ float bf2f(unsigned short h) {
    return __uint_as_float(((unsigned int)h) << 16);
}

// ---------------- small elementwise kernels --------------------------
__global__ void k_cvt_bf16(const float* __restrict__ in,
                           unsigned short* __restrict__ out, int n) {
    int i = blockIdx.x * blockDim.x + threadIdx.x;
    if (i < n) out[i] = f2bf(in[i]);
}

__global__ void k_degree(const int* __restrict__ dst, float* __restrict__ cnt) {
    int e = blockIdx.x * blockDim.x + threadIdx.x;
    if (e < EE) atomicAdd(&cnt[dst[e]], 1.0f);
}

// layer-1 scatter: agg[dst, f..f+3] += x[src, f..f+3]   (f32 source, D=128)
__global__ void k_scatter_f32(const int* __restrict__ src, const int* __restrict__ dst,
                              const float* __restrict__ x, float* __restrict__ agg) {
    int tid = blockIdx.x * blockDim.x + threadIdx.x;   // EE * 32 threads
    int e = tid >> 5;
    int f = (tid & 31) << 2;
    if (e >= EE) return;
    v4f v = *(const v4f*)(x + (size_t)src[e] * DIN + f);
    float* a = agg + (size_t)dst[e] * DIN + f;
    atomicAdd(a + 0, v.x); atomicAdd(a + 1, v.y);
    atomicAdd(a + 2, v.z); atomicAdd(a + 3, v.w);
}

// layer-2 scatter: agg[dst, f..f+3] += h[src, f..f+3]   (bf16 source, D=256)
__global__ void k_scatter_bf16(const int* __restrict__ src, const int* __restrict__ dst,
                               const unsigned short* __restrict__ hb, float* __restrict__ agg) {
    int tid = blockIdx.x * blockDim.x + threadIdx.x;   // EE * 64 threads
    int e = tid >> 6;
    int f = (tid & 63) << 2;
    if (e >= EE) return;
    v4us v = *(const v4us*)(hb + (size_t)src[e] * DHID + f);
    float* a = agg + (size_t)dst[e] * DHID + f;
    atomicAdd(a + 0, bf2f(v.x)); atomicAdd(a + 1, bf2f(v.y));
    atomicAdd(a + 2, bf2f(v.z)); atomicAdd(a + 3, bf2f(v.w));
}

// mean = agg / max(cnt,1), emitted as bf16.  KSHIFT = log2(feature dim)
template <int KSHIFT>
__global__ void k_mean_bf16(const float* __restrict__ agg, const float* __restrict__ cnt,
                            unsigned short* __restrict__ mb, int total) {
    int i = blockIdx.x * blockDim.x + threadIdx.x;
    if (i >= total) return;
    int n = i >> KSHIFT;
    float r = 1.0f / fmaxf(cnt[n], 1.0f);
    mb[i] = f2bf(agg[i] * r);
}

// ---------------- WMMA fragment loads (bf16, 16x16x32) ---------------
// A (16xK slab, row-major MxK): lane holds row m, K-groups
//   [k0 + 8*hi, +8) and [k0 + 16 + 8*hi, +8)
static __device__ __forceinline__ v16bf load_a_frag(const unsigned short* __restrict__ A,
                                                    int K, int m, int k0, int hi) {
    const unsigned short* row = A + (size_t)m * K + k0 + (hi << 3);
    union { v8us h[2]; v16us u; } t;
    t.h[0] = *(const v8us*)(row);
    t.h[1] = *(const v8us*)(row + 16);
    return __builtin_bit_cast(v16bf, t.u);
}
// B = W^T (Kx16): lane holds output column n -> contiguous row n of W,
//   K-groups [k0 + 8*hi, +8) and [k0 + 16 + 8*hi, +8)
static __device__ __forceinline__ v16bf load_b_frag(const unsigned short* __restrict__ W,
                                                    int K, int n, int k0, int hi) {
    const unsigned short* row = W + (size_t)n * K + k0 + (hi << 3);
    union { v8us h[2]; v16us u; } t;
    t.h[0] = *(const v8us*)(row);
    t.h[1] = *(const v8us*)(row + 16);
    return __builtin_bit_cast(v16bf, t.u);
}

// ---------------- fused SAGE GEMM: D = [relu](Am@Wl^T + Ar@Wr^T + b) --
// One wave computes a 16 x (16*NTPW) output slab: the A fragment pair is
// loaded once per k-step and reused across NTPW n-tiles (register tiling),
// so per k-step: 4 A-loads + 4*NTPW B-loads feed 2*NTPW WMMAs.
template <int K, int NOUT, int NTPW, bool RELU, bool BF16OUT>
__global__ void k_sage_gemm(const unsigned short* __restrict__ Am,
                            const unsigned short* __restrict__ Ar,
                            const unsigned short* __restrict__ Wl,
                            const unsigned short* __restrict__ Wr,
                            const float* __restrict__ bias,
                            float* __restrict__ outf,
                            unsigned short* __restrict__ outb) {
    const int NT = NOUT / 16;
    const int MT = NN / 16;                   // 6250, exact
    const int GPR = NT / NTPW;                // n-tile groups per m-tile
    int wave = threadIdx.x >> 5;
    int lane = threadIdx.x & 31;
    int grp = blockIdx.x * (blockDim.x >> 5) + wave;
    if (grp >= MT * GPR) return;              // whole-wave exit; EXEC all-1s for WMMA
    int mt = grp / GPR;
    int n0 = (grp % GPR) * NTPW * 16;
    int hi = lane >> 4;
    int lm = lane & 15;
    int m = mt * 16 + lm;                     // A row handled by this lane

    v8f acc[NTPW];
#pragma unroll
    for (int j = 0; j < NTPW; ++j) acc[j] = (v8f){};

#pragma unroll
    for (int k0 = 0; k0 < K; k0 += 32) {
        v16bf am = load_a_frag(Am, K, m, k0, hi);
        v16bf ar = load_a_frag(Ar, K, m, k0, hi);
        if (k0 + 32 < K) {                    // hint next A k-slab into cache
            __builtin_prefetch(Am + (size_t)m * K + k0 + 32, 0, 1);
            __builtin_prefetch(Ar + (size_t)m * K + k0 + 32, 0, 1);
        }
#pragma unroll
        for (int j = 0; j < NTPW; ++j) {
            int n = n0 + j * 16 + lm;
            v16bf bl = load_b_frag(Wl, K, n, k0, hi);
            v16bf br = load_b_frag(Wr, K, n, k0, hi);
            acc[j] = __builtin_amdgcn_wmma_f32_16x16x32_bf16(false, am, false, bl,
                                                             (short)0, acc[j], false, false);
            acc[j] = __builtin_amdgcn_wmma_f32_16x16x32_bf16(false, ar, false, br,
                                                             (short)0, acc[j], false, false);
        }
    }

    int mbase = mt * 16 + (hi << 3);          // C/D: VGPR r -> row mbase + r
#pragma unroll
    for (int j = 0; j < NTPW; ++j) {
        int n = n0 + j * 16 + lm;
        float bv = bias[n];
#pragma unroll
        for (int r = 0; r < 8; ++r) {
            float v = acc[j][r] + bv;
            if (RELU) v = fmaxf(v, 0.0f);
            size_t off = (size_t)(mbase + r) * NOUT + n;
            if (BF16OUT) outb[off] = f2bf(v);
            else         outf[off] = v;
        }
    }
}

// ---------------- host-side orchestration ----------------------------
static inline size_t align_up(size_t x) { return (x + 1023) & ~(size_t)1023; }

extern "C" void kernel_launch(void* const* d_in, const int* in_sizes, int n_in,
                              void* d_out, int out_size, void* d_ws, size_t ws_size,
                              hipStream_t stream) {
    (void)in_sizes; (void)n_in; (void)out_size; (void)ws_size;
    const float* x    = (const float*)d_in[0];
    const int*   eidx = (const int*)d_in[1];
    const int*   src  = eidx;
    const int*   dst  = eidx + EE;
    const float* W1l  = (const float*)d_in[2];
    const float* b1   = (const float*)d_in[3];
    const float* W1r  = (const float*)d_in[4];
    const float* W2l  = (const float*)d_in[5];
    const float* b2   = (const float*)d_in[6];
    const float* W2r  = (const float*)d_in[7];
    float* out = (float*)d_out;

    // workspace carve-up (~231 MB)
    char* ws = (char*)d_ws;
    size_t o = 0;
    float* cnt = (float*)(ws + o);          o = align_up(o + (size_t)NN * 4);
    float* agg = (float*)(ws + o);          o = align_up(o + (size_t)NN * DHID * 4);
    unsigned short* xb  = (unsigned short*)(ws + o); o = align_up(o + (size_t)NN * DIN * 2);
    unsigned short* mb  = (unsigned short*)(ws + o); o = align_up(o + (size_t)NN * DHID * 2);
    unsigned short* hb  = (unsigned short*)(ws + o); o = align_up(o + (size_t)NN * DHID * 2);
    unsigned short* w1l = (unsigned short*)(ws + o); o = align_up(o + (size_t)DHID * DIN * 2);
    unsigned short* w1r = (unsigned short*)(ws + o); o = align_up(o + (size_t)DHID * DIN * 2);
    unsigned short* w2l = (unsigned short*)(ws + o); o = align_up(o + (size_t)DOUT * DHID * 2);
    unsigned short* w2r = (unsigned short*)(ws + o); o = align_up(o + (size_t)DOUT * DHID * 2);

    const int B = 256;

    // zero accumulators (memset nodes are graph-capturable)
    hipMemsetAsync(cnt, 0, (size_t)NN * 4, stream);
    hipMemsetAsync(agg, 0, (size_t)NN * DHID * 4, stream);

    // bf16 conversions
    k_cvt_bf16<<<(NN * DIN + B - 1) / B, B, 0, stream>>>(x, xb, NN * DIN);
    k_cvt_bf16<<<(DHID * DIN + B - 1) / B, B, 0, stream>>>(W1l, w1l, DHID * DIN);
    k_cvt_bf16<<<(DHID * DIN + B - 1) / B, B, 0, stream>>>(W1r, w1r, DHID * DIN);
    k_cvt_bf16<<<(DOUT * DHID + B - 1) / B, B, 0, stream>>>(W2l, w2l, DOUT * DHID);
    k_cvt_bf16<<<(DOUT * DHID + B - 1) / B, B, 0, stream>>>(W2r, w2r, DOUT * DHID);

    // degrees (shared by both layers)
    k_degree<<<(EE + B - 1) / B, B, 0, stream>>>(dst, cnt);

    // ---- layer 1 ----
    k_scatter_f32<<<(EE * 32) / B, B, 0, stream>>>(src, dst, x, agg);
    k_mean_bf16<7><<<(NN * DIN + B - 1) / B, B, 0, stream>>>(agg, cnt, mb, NN * DIN);
    {   // h = relu(mean@W1l^T + x@W1r^T + b1) -> bf16
        int groups = (NN / 16) * ((DHID / 16) / 4);   // NTPW = 4
        k_sage_gemm<DIN, DHID, 4, true, true>
            <<<(groups + 7) / 8, B, 0, stream>>>(mb, xb, w1l, w1r, b1, nullptr, hb);
    }

    // ---- layer 2 ----
    hipMemsetAsync(agg, 0, (size_t)NN * DHID * 4, stream);
    k_scatter_bf16<<<(EE * 64) / B, B, 0, stream>>>(src, dst, hb, agg);
    k_mean_bf16<8><<<(NN * DHID + B - 1) / B, B, 0, stream>>>(agg, cnt, mb, NN * DHID);
    {   // out = mean@W2l^T + h@W2r^T + b2 -> f32  (one wave per 16x64 slab)
        int groups = (NN / 16) * ((DOUT / 16) / 4);   // NTPW = 4
        k_sage_gemm<DHID, DOUT, 4, false, false>
            <<<(groups + 7) / 8, B, 0, stream>>>(mb, hb, w2l, w2r, b2, out, nullptr);
    }
}
